// TopkPointExtractor_74294344286811
// MI455X (gfx1250) — matrix operand order repaired and checked by reading, not compiled, not run on hardware
//
#include <hip/hip_runtime.h>
#include <hip/hip_bf16.h>

typedef __attribute__((ext_vector_type(16))) _Float16 v16h;
typedef __attribute__((ext_vector_type(8)))  float    v8f;

// TDM descriptor vector types (probe-verified builtin signatures)
typedef unsigned int tdm_v4u __attribute__((ext_vector_type(4)));
typedef int          tdm_v8i __attribute__((ext_vector_type(8)));
typedef int          tdm_v4i __attribute__((ext_vector_type(4)));

#define BQ     16
#define HH     128
#define WW     128
#define HW     (HH*WW)          // 16384
#define PP     256
#define CAP    1024
#define LCACHE 8192             // floats cached in LDS by the TDM (32 KB)

// ---------------- init per-batch stats --------------------------------------
__global__ void k_init(unsigned* mnb, unsigned* mxb, unsigned* pmb) {
    int t = threadIdx.x;
    if (t < BQ) { mnb[t] = 0x7F800000u; mxb[t] = 0u; pmb[t] = 0x7F800000u; }
}

// ---------------- conv1x1 (WMMA) + sigmoid + per-batch min/max --------------
// One wave computes 16 consecutive spatial positions: A = 16x32 f16 feat tile,
// B = weight vector replicated into all 16 columns. Two WMMAs cover K=64.
__global__ __launch_bounds__(256) void k_conv(const float* __restrict__ feat,
                                              const float* __restrict__ wconv,
                                              const float* __restrict__ bconv,
                                              float* __restrict__ sig,
                                              unsigned* mnb, unsigned* mxb) {
    int lane = threadIdx.x & 31;
    int wv   = threadIdx.x >> 5;
    int gw   = blockIdx.x * 8 + wv;        // global wave id
    int posBase = gw * 16;                 // [0, B*HW)
    int b       = posBase >> 14;
    int posInB  = posBase & (HW - 1);      // HW divisible by 16 -> no straddle
    int m  = lane & 15;
    int hi = lane >> 4;

    const float* fb = feat + (size_t)b * 64 * HW + posInB + m;

    // A layout (16-bit A 16x32): lanes 0-15: K 0-7 & 16-23; lanes 16-31: K 8-15 & 24-31
    v16h a0, a1, bm0, bm1;
#pragma unroll
    for (int v = 0; v < 8; ++v) {
        int kp = (v < 4) ? (hi * 8 + 2 * v) : (16 + hi * 8 + 2 * (v - 4));
        a0[2 * v]     = (_Float16)fb[(size_t)kp * HW];
        a0[2 * v + 1] = (_Float16)fb[(size_t)(kp + 1) * HW];
        a1[2 * v]     = (_Float16)fb[(size_t)(kp + 32) * HW];
        a1[2 * v + 1] = (_Float16)fb[(size_t)(kp + 33) * HW];
    }
    // B layout (32x16): lanes 0-15 hold K 0-7 & 16-23, lanes 16-31 hold K 8-15 & 24-31;
    // every column is the same weight vector.
#pragma unroll
    for (int e = 0; e < 16; ++e) {
        int k = (e < 8) ? (hi * 8 + e) : (16 + hi * 8 + (e - 8));
        bm0[e] = (_Float16)wconv[k];
        bm1[e] = (_Float16)wconv[k + 32];
    }

    v8f c = {};
    c = __builtin_amdgcn_wmma_f32_16x16x32_f16(false, a0, false, bm0, (short)0, c, false, false);
    c = __builtin_amdgcn_wmma_f32_16x16x32_f16(false, a1, false, bm1, (short)0, c, false, false);

    // D layout: lane(hi,n): d[v] = (M = v + 8*hi, N = n). Writers: n<8, M = 8*hi+n.
    int n = lane & 15;
    if (n < 8) {
        float val = c[n] + bconv[0];
        float s = __builtin_amdgcn_rcpf(1.0f + __expf(-val));   // v_rcp_f32
        int pos = posInB + hi * 8 + n;
        sig[(size_t)b * HW + pos] = s;
        unsigned ub = __float_as_uint(s);      // s in (0,1): bit order == float order
        atomicMin(&mnb[b], ub);
        atomicMax(&mxb[b], ub);
    }
}

// ---------------- 4x4 block-max pool of img[:,0] + per-batch spatial min ----
__global__ void k_pool(const float* __restrict__ img, float* __restrict__ pooled,
                       unsigned* pmb) {
    int t = blockIdx.x * blockDim.x + threadIdx.x;   // B*HW threads
    int b = t >> 14;
    int cell = t & (HW - 1);
    int cy = cell >> 7, cx = cell & 127;
    const float* base = img + (size_t)b * 3 * 512 * 512 + (size_t)(cy * 4) * 512 + cx * 4;
    __builtin_prefetch(base + 4 * 512, 0, 3);        // global_prefetch_b8
    float mx = -1e30f;
#pragma unroll
    for (int r = 0; r < 4; ++r) {
        float4 v = *(const float4*)(base + (size_t)r * 512);
        mx = fmaxf(mx, fmaxf(fmaxf(v.x, v.y), fmaxf(v.z, v.w)));
    }
    pooled[t] = mx;
    atomicMin(&pmb[b], __float_as_uint(mx));         // mx >= 0
}

// ---------------- x = norm(sigmoid) * (pooled - pmin); S = sum_b x ----------
__global__ void k_combine(float* __restrict__ x, const float* __restrict__ pooled,
                          float* __restrict__ S, const unsigned* mnb,
                          const unsigned* mxb, const unsigned* pmb) {
    int pos = blockIdx.x * blockDim.x + threadIdx.x;  // HW threads
    float acc = 0.f;
#pragma unroll
    for (int b = 0; b < BQ; ++b) {
        float mn = __uint_as_float(mnb[b]);
        float mx = __uint_as_float(mxb[b]);
        float pm = __uint_as_float(pmb[b]);
        float inv = __builtin_amdgcn_rcpf(mx - mn);
        float s  = x[(size_t)b * HW + pos];
        float v  = ((s - mn) * inv) * (pooled[(size_t)b * HW + pos] - pm);
        x[(size_t)b * HW + pos] = v;
        acc += v;
    }
    S[pos] = acc;
}

// ---------------- exact top-256 per batch: radix-select + bitonic sort ------
// The first 32 KB of this batch's value slice is DMA'd into LDS once by the
// Tensor Data Mover (tracked by TENSORcnt); the radix passes then read half
// from LDS and half from L2.
__global__ __launch_bounds__(1024) void k_topk(const float* __restrict__ x,
                                               float* __restrict__ topv,
                                               int* __restrict__ topi) {
    __shared__ __align__(16) unsigned smx[LCACHE];     // 32 KB TDM destination
    __shared__ unsigned long long cand[CAP];           // 8 KB
    __shared__ unsigned hist[256];
    __shared__ unsigned sh_prefix;
    __shared__ int sh_need;
    __shared__ int sh_cnt;

    int b = blockIdx.x;
    int tid = threadIdx.x;
    const float* xb = x + (size_t)b * HW;

#if __has_builtin(__builtin_amdgcn_tensor_load_to_lds)
    if (tid < 32) {                     // wave 0 issues the TDM (EXEC ignored)
        unsigned lds = (unsigned)(size_t)&smx[0];             // LDS byte offset
        unsigned long long ga = (unsigned long long)(size_t)xb;
        // D# group 0: count=1, lds_addr, global_addr[56:0], type=2
        tdm_v4u g0 = { 1u, lds, (unsigned)ga,
                       (unsigned)((ga >> 32) & 0x01FFFFFFu) | (2u << 30) };
        // D# group 1: data_size=4B; tensor_dim0=tile_dim0=8192; dim1=1; strides
        tdm_v8i g1 = { 0x00020000,                       // data_size=2 (4B)
                       (int)((LCACHE & 0xFFFF) << 16),   // tensor_dim0 lo16
                       (int)((LCACHE >> 16) | (1 << 16)),// dim0 hi | tensor_dim1 lo
                       (int)(LCACHE << 16),              // dim1 hi | tile_dim0
                       1,                                // tile_dim1=1, tile_dim2=0
                       LCACHE,                           // tensor_dim0_stride lo32
                       (int)((LCACHE & 0xFFFF) << 16),   // stride0 hi | stride1 lo
                       0 };                              // stride1 hi
        tdm_v4i gz = { 0, 0, 0, 0 };
#if __clang_major__ >= 23
        tdm_v8i gz8 = { 0, 0, 0, 0, 0, 0, 0, 0 };
        __builtin_amdgcn_tensor_load_to_lds(g0, g1, gz, gz, gz8, 0);
#else
        __builtin_amdgcn_tensor_load_to_lds(g0, g1, gz, gz, 0);
#endif
        __builtin_amdgcn_s_wait_tensorcnt(0);
    }
    __syncthreads();
#else
    for (int i = tid; i < LCACHE; i += 1024) smx[i] = __float_as_uint(xb[i]);
    __syncthreads();
#endif

    if (tid == 0) { sh_prefix = 0u; sh_need = PP; }
    __syncthreads();

    // 4 x 8-bit radix passes over float bits (all values >= 0 -> monotone)
    for (int pass = 0; pass < 4; ++pass) {
        for (int i = tid; i < 256; i += 1024) hist[i] = 0;
        __syncthreads();
        unsigned prefix = sh_prefix;
        int shift = 24 - 8 * pass;
        unsigned hiMask = (pass == 0) ? 0u : (0xFFFFFFFFu << (shift + 8));
        for (int i = tid; i < HW; i += 1024) {
            unsigned u = (i < LCACHE) ? smx[i] : __float_as_uint(xb[i]);
            if ((u & hiMask) == (prefix & hiMask))
                atomicAdd(&hist[(u >> shift) & 255], 1u);
        }
        __syncthreads();
        if (tid == 0) {
            int need = sh_need;
            unsigned cum = 0;
            int bsel = 0;
            for (int bin = 255; bin >= 0; --bin) {
                unsigned c = hist[bin];
                if (cum + c >= (unsigned)need) { bsel = bin; break; }
                cum += c;
            }
            sh_prefix = prefix | ((unsigned)bsel << shift);
            sh_need = need - (int)cum;
        }
        __syncthreads();
    }
    unsigned T = sh_prefix;

    // collect candidates (all >= T), packed key = (valbits<<32) | ~idx
    if (tid == 0) sh_cnt = 0;
    for (int i = tid; i < CAP; i += 1024) cand[i] = 0ull;
    __syncthreads();
    for (int i = tid; i < HW; i += 1024) {
        unsigned u = (i < LCACHE) ? smx[i] : __float_as_uint(xb[i]);
        if (u >= T) {
            int slot = atomicAdd(&sh_cnt, 1);
            if (slot < CAP)
                cand[slot] = ((unsigned long long)u << 32) | (unsigned)(~i);
        }
    }
    __syncthreads();

    // bitonic sort descending: value desc, index asc (matches jax top_k ties)
    for (unsigned k = 2; k <= CAP; k <<= 1) {
        for (unsigned j = k >> 1; j > 0; j >>= 1) {
            unsigned i = tid, ixj = i ^ j;
            if (ixj > i) {
                unsigned long long a = cand[i], c = cand[ixj];
                bool asc = ((i & k) != 0);
                if (asc ? (a > c) : (a < c)) { cand[i] = c; cand[ixj] = a; }
            }
            __syncthreads();
        }
    }

    if (tid < PP) {
        unsigned long long key = cand[tid];
        topv[b * PP + tid] = __uint_as_float((unsigned)(key >> 32));
        topi[b * PP + tid] = (int)(unsigned)(~(unsigned)key);
    }
}

// ---------------- soft coordinate refinement --------------------------------
__global__ void k_points(const float* __restrict__ x, const float* __restrict__ S,
                         const int* __restrict__ topi, float* __restrict__ out) {
    int t = blockIdx.x * blockDim.x + threadIdx.x;   // B*PP threads
    int b = t >> 8, p = t & (PP - 1);
    int idx = topi[b * PP + p];
    int ax = idx & 127, oy = idx >> 7;

    float den = 0.f, sa = 0.f, so = 0.f, sv = 0.f;
#pragma unroll
    for (int dy = -1; dy <= 1; ++dy) {
#pragma unroll
        for (int dx = -1; dx <= 1; ++dx) {
            int wq = 2 - (dy < 0 ? -dy : dy) - (dx < 0 ? -dx : dx);
            if (wq <= 0) continue;                    // diagonals have weight 0
            int ay = oy + dy, axp = ax + dx;
            if (ay < 0 || ay >= HH || axp < 0 || axp >= WW) continue;
            int q = ay * WW + axp;
            float wsv = (float)wq * S[q];
            float xg  = x[(size_t)b * HW + q];
            den += wsv;
            sa  += wsv * (float)axp;
            so  += wsv * (float)ay;
            sv  += wsv * xg;
        }
    }
    float rden = __builtin_amdgcn_rcpf(den);
    float* o = out + (size_t)(b * PP + p) * 4;
    o[0] = sa * rden;
    o[1] = so * rden;
    o[2] = 0.f;
    o[3] = sv * rden;
}

extern "C" void kernel_launch(void* const* d_in, const int* in_sizes, int n_in,
                              void* d_out, int out_size, void* d_ws, size_t ws_size,
                              hipStream_t stream) {
    (void)in_sizes; (void)n_in; (void)out_size; (void)ws_size;
    const float* img   = (const float*)d_in[0];   // [16,3,512,512]
    const float* feat  = (const float*)d_in[1];   // [16,64,128,128]
    const float* wconv = (const float*)d_in[2];   // [64]
    const float* bconv = (const float*)d_in[3];   // [1]
    float* out = (float*)d_out;                   // [16,256,4]

    // workspace layout (floats)
    float* sig    = (float*)d_ws;                 // B*HW (sigmoid, then x)
    float* pooled = sig + (size_t)BQ * HW;        // B*HW
    float* S      = pooled + (size_t)BQ * HW;     // HW
    unsigned* mnb = (unsigned*)(S + HW);
    unsigned* mxb = mnb + BQ;
    unsigned* pmb = mxb + BQ;
    float* topv   = (float*)(pmb + BQ);           // B*PP
    int*   topi   = (int*)(topv + BQ * PP);       // B*PP

    k_init<<<1, 64, 0, stream>>>(mnb, mxb, pmb);
    k_conv<<<(BQ * HW) / 128, 256, 0, stream>>>(feat, wconv, bconv, sig, mnb, mxb);
    k_pool<<<(BQ * HW) / 256, 256, 0, stream>>>(img, pooled, pmb);
    k_combine<<<HW / 256, 256, 0, stream>>>(sig, pooled, S, mnb, mxb, pmb);
    k_topk<<<BQ, 1024, 0, stream>>>(sig, topv, topi);
    k_points<<<(BQ * PP) / 256, 256, 0, stream>>>(sig, S, topi, out);
}
